// GAT_LSTM_Attn_Model_31662498906485
// MI455X (gfx1250) — matrix operand order repaired and checked by reading, not compiled, not run on hardware
//
#include <hip/hip_runtime.h>
#include <hip/hip_bf16.h>

// ---------------------------------------------------------------------------
// Model dims
// ---------------------------------------------------------------------------
#define BB    256
#define SS    64
#define NN    17
#define CIN   3
#define HID   64
#define HEADS 2
#define OUT_  128
#define NCLS  10
#define GN    (BB * SS * NN)       // 278528 GEMM rows for GAT (multiple of 32)
#define NG    (BB * SS)            // 16384 graphs (multiple of 32)
#define EDG   81                   // 64 edges + 17 self loops
#define LSTM_IN (NN * HEADS * HID) // 2176

typedef __attribute__((ext_vector_type(16))) __bf16 v16bf;
typedef __attribute__((ext_vector_type(8)))  float  v8f;

// ---------------------------------------------------------------------------
// Generic fp32-in / fp32-out GEMM with bf16 WMMA core.
//   C[M,N] = act( A[M,K] @ op(B) + bias1 + bias2 )
//   transB==0 : B is [K,N] row-major (ldb = N)
//   transB==1 : B is [N,K] row-major (ldb = K)  (i.e. A @ B^T)
// Requirements (guaranteed by all call sites): M % 32 == 0, N % 128 == 0.
// Block: 256 threads = 8 waves. Block tile 32(M) x 128(N); wave w owns
// n-subtile w and both m-subtiles -> 2 WMMA per K-step, shared B fragment.
// ---------------------------------------------------------------------------
__global__ __launch_bounds__(256)
void gemm_bf16_wmma(const float* __restrict__ A, int lda,
                    const float* __restrict__ B, int ldb, int transB,
                    const float* __restrict__ bias1, const float* __restrict__ bias2,
                    float* __restrict__ C, int ldc,
                    int M, int K, int N, int relu)
{
    __shared__ __bf16 As[32][32];   // 2 KB
    __shared__ __bf16 Bs[128][32];  // 8 KB

    const int tid  = threadIdx.x;
    const int wave = tid >> 5;
    const int lane = tid & 31;
    const int m0   = blockIdx.x * 32;
    const int n0   = blockIdx.y * 128;

    const int half = lane >> 4;                 // 0: lanes 0-15, 1: lanes 16-31
    const int mr   = lane & 15;                 // A row within 16-row subtile
    const int nc   = (lane & 15) + wave * 16;   // B column within 128-wide tile

    v8f acc0 = {};
    v8f acc1 = {};

    const bool fast = ((K & 31) == 0) && ((lda & 3) == 0) && ((ldb & 3) == 0);

    for (int k0 = 0; k0 < K; k0 += 32) {
        if (fast) {
            // ---- A tile 32x32: one float4 per thread, unguarded ----
            {
                const int i  = tid << 2;                  // 0..1020
                const int r  = i >> 5;                    // 0..31
                const int kk = i & 31;                    // multiple of 4
                const float* ap = A + (size_t)(m0 + r) * lda + (k0 + kk);
                const float4 v = *(const float4*)ap;
                As[r][kk]     = (__bf16)v.x;
                As[r][kk + 1] = (__bf16)v.y;
                As[r][kk + 2] = (__bf16)v.z;
                As[r][kk + 3] = (__bf16)v.w;
                if (k0 + 32 < K) __builtin_prefetch(ap + 32, 0, 3);
            }
            // ---- B tile 32x128 (n-major in LDS): 4 float4 per thread ----
#pragma unroll
            for (int j = 0; j < 4; ++j) {
                const int i = (tid << 2) + (j << 10);     // 0..4092
                if (transB) {
                    const int n  = i >> 5;                // 0..127
                    const int kk = i & 31;                // multiple of 4
                    const float* bp = B + (size_t)(n0 + n) * ldb + (k0 + kk);
                    const float4 v = *(const float4*)bp;
                    Bs[n][kk]     = (__bf16)v.x;
                    Bs[n][kk + 1] = (__bf16)v.y;
                    Bs[n][kk + 2] = (__bf16)v.z;
                    Bs[n][kk + 3] = (__bf16)v.w;
                    if (k0 + 32 < K) __builtin_prefetch(bp + 32, 0, 3);
                } else {
                    const int kk = i >> 7;                // 0..31
                    const int n  = i & 127;               // multiple of 4
                    const float* bp = B + (size_t)(k0 + kk) * ldb + (n0 + n);
                    const float4 v = *(const float4*)bp;
                    Bs[n][kk]     = (__bf16)v.x;
                    Bs[n + 1][kk] = (__bf16)v.y;
                    Bs[n + 2][kk] = (__bf16)v.z;
                    Bs[n + 3][kk] = (__bf16)v.w;
                    if (k0 + 32 < K) __builtin_prefetch(bp + 32 * ldb, 0, 3);
                }
            }
        } else {
            // ---- guarded slow path (only layer 0: K=3, lda=3) ----
            for (int i = tid; i < 32 * 32; i += 256) {
                const int r = i >> 5, kk = i & 31;
                const int gk = k0 + kk;
                float v = 0.f;
                if (gk < K) v = A[(size_t)(m0 + r) * lda + gk];
                As[r][kk] = (__bf16)v;
            }
            for (int i = tid; i < 128 * 32; i += 256) {
                const int n = i >> 5, kk = i & 31;
                const int gk = k0 + kk;
                float v = 0.f;
                if (gk < K)
                    v = transB ? B[(size_t)(n0 + n) * ldb + gk]
                               : B[(size_t)gk * ldb + (n0 + n)];
                Bs[n][kk] = (__bf16)v;
            }
        }
        __syncthreads();

        // ---- fragments per ISA 7.12.2 (16-bit A 16x32 / B 32x16) ----
        v16bf av0, av1, bv;
#pragma unroll
        for (int v = 0; v < 8; ++v) {
            const int kbA = ((v & 4) << 2) + half * 8 + ((v & 3) << 1); // v>=4 -> +16
            const int kbB = ((v & 4) << 2) + half * 8 + ((v & 3) << 1);
            av0[2 * v]     = As[mr][kbA];
            av0[2 * v + 1] = As[mr][kbA + 1];
            av1[2 * v]     = As[mr + 16][kbA];
            av1[2 * v + 1] = As[mr + 16][kbA + 1];
            bv[2 * v]      = Bs[nc][kbB];
            bv[2 * v + 1]  = Bs[nc][kbB + 1];
        }
        acc0 = __builtin_amdgcn_wmma_f32_16x16x32_bf16(
            false, av0, false, bv, (short)0, acc0, false, false);
        acc1 = __builtin_amdgcn_wmma_f32_16x16x32_bf16(
            false, av1, false, bv, (short)0, acc1, false, false);
        __syncthreads();
    }

    // ---- epilogue: C layout lane -> n = lane&15, VGPR r -> m = r + half*8 ----
    const int gn = n0 + nc;
    float bb = 0.f;
    if (bias1) bb += bias1[gn];
    if (bias2) bb += bias2[gn];
#pragma unroll
    for (int r = 0; r < 8; ++r) {
        const int gm = m0 + r + half * 8;
        float v0 = acc0[r] + bb;
        float v1 = acc1[r] + bb;
        if (relu) { v0 = fmaxf(v0, 0.f); v1 = fmaxf(v1, 0.f); }
        C[(size_t)gm * ldc + gn]        = v0;
        C[(size_t)(gm + 16) * ldc + gn] = v1;
    }
}

// ---------------------------------------------------------------------------
// GAT attention + aggregation (in-place on h), one 128-thread block per graph.
// The 17x128 fp32 tile is staged global->LDS with CDNA5 async copies
// (global_load_async_to_lds_b128, fenced by s_wait_asynccnt).
// ---------------------------------------------------------------------------
__global__ __launch_bounds__(128)
void gat_aggregate(float* __restrict__ h, const int* __restrict__ ei,
                   const float* __restrict__ a_src, const float* __restrict__ a_dst,
                   const float* __restrict__ bias, int relu)
{
    const int g   = blockIdx.x;
    const int tid = threadIdx.x;

    __shared__ float hs[NN][128];
    __shared__ float outa[NN][128];
    __shared__ float as_[NN][2], ad_[NN][2];
    __shared__ float nmax[NN][2], nsum[NN][2];
    __shared__ int   se[EDG], de[EDG];

    float* hg = h + (size_t)g * NN * 128;

    // ---- async copy of the 17x128 tile: 544 x 16B chunks ----
#ifdef __HIP_DEVICE_COMPILE__
    {
        const unsigned ldsbase = (unsigned)(size_t)&hs[0][0];
        for (int i = tid; i < (NN * 128) / 4; i += 128) {
            const unsigned loff = ldsbase + (unsigned)i * 16u;
            const float* gp = hg + (size_t)i * 4;
            asm volatile("global_load_async_to_lds_b128 %0, %1, off"
                         :: "v"(loff), "v"(gp) : "memory");
        }
        asm volatile("s_wait_asynccnt 0x0" ::: "memory");
    }
#else
    for (int i = tid; i < NN * 128; i += 128) hs[i >> 7][i & 127] = hg[i];
#endif
    for (int i = tid; i < NN * 128; i += 128) outa[i >> 7][i & 127] = 0.f;
    if (tid < 64)       { se[tid] = ei[tid]; de[tid] = ei[64 + tid]; }
    else if (tid < EDG) { se[tid] = tid - 64; de[tid] = tid - 64; }   // self loops
    __syncthreads();

    // per-(node, head) attention dots
    if (tid < NN * HEADS) {
        const int n = tid >> 1, hd = tid & 1;
        float s = 0.f, d = 0.f;
        for (int k = 0; k < HID; ++k) {
            const float v = hs[n][hd * HID + k];
            s = fmaf(v, a_src[hd * HID + k], s);
            d = fmaf(v, a_dst[hd * HID + k], d);
        }
        as_[n][hd] = s; ad_[n][hd] = d;
    }
    __syncthreads();

    // per-(dst, head) max + exp-sum over incident edges
    if (tid < NN * HEADS) {
        const int n = tid >> 1, hd = tid & 1;
        float mx = -1e30f;
        for (int e = 0; e < EDG; ++e) {
            if (de[e] == n) {
                float ev = as_[se[e]][hd] + ad_[n][hd];
                ev = ev > 0.f ? ev : 0.2f * ev;
                mx = fmaxf(mx, ev);
            }
        }
        float sm = 0.f;
        for (int e = 0; e < EDG; ++e) {
            if (de[e] == n) {
                float ev = as_[se[e]][hd] + ad_[n][hd];
                ev = ev > 0.f ? ev : 0.2f * ev;
                sm += __expf(ev - mx);
            }
        }
        nmax[n][hd] = mx; nsum[n][hd] = sm;
    }
    __syncthreads();

    // aggregation: thread tid owns output column tid (head = tid/64)
    const int hd = tid >> 6;
    for (int e = 0; e < EDG; ++e) {
        const int s = se[e], d = de[e];
        float ev = as_[s][hd] + ad_[d][hd];
        ev = ev > 0.f ? ev : 0.2f * ev;
        const float alpha = __expf(ev - nmax[d][hd]) / (nsum[d][hd] + 1e-16f);
        outa[d][tid] = fmaf(alpha, hs[s][tid], outa[d][tid]);
    }

    const float bb = bias[tid];
    for (int n = 0; n < NN; ++n) {
        float v = outa[n][tid] + bb;
        if (relu) v = fmaxf(v, 0.f);
        hg[n * 128 + tid] = v;
    }
}

// ---------------------------------------------------------------------------
// LSTM recurrence. X = precomputed x@Wih^T + bih + bhh, [B, T, 512].
// One block per batch row; thread j owns hidden unit j. Whh (256 KB) streams
// from L2 each step. Gate order i, f, g, o.
// ---------------------------------------------------------------------------
__global__ __launch_bounds__(128)
void lstm_recur(const float* __restrict__ X, const float* __restrict__ whh,
                float* __restrict__ H, int T)
{
    const int b = blockIdx.x;
    const int j = threadIdx.x;
    __shared__ float hs[128];
    hs[j] = 0.f;
    float c = 0.f;
    __syncthreads();

    for (int t = 0; t < T; ++t) {
        const float* xb = X + ((size_t)b * T + t) * 512;
        float gpre[4];
#pragma unroll
        for (int k = 0; k < 4; ++k) {
            float a = xb[k * 128 + j];
            const float* wr = whh + (size_t)(k * 128 + j) * 128;
#pragma unroll 8
            for (int m = 0; m < 128; ++m) a = fmaf(wr[m], hs[m], a);
            gpre[k] = a;
        }
        __syncthreads();
        const float ig = 1.f / (1.f + __expf(-gpre[0]));
        const float fg = 1.f / (1.f + __expf(-gpre[1]));
        const float gg = tanhf(gpre[2]);
        const float og = 1.f / (1.f + __expf(-gpre[3]));
        c = fg * c + ig * gg;
        const float hv = og * tanhf(c);
        hs[j] = hv;
        H[((size_t)b * T + t) * 128 + j] = hv;
        __syncthreads();
    }
}

// ---------------------------------------------------------------------------
// Temporal attention pooling + final FC. One block per batch row.
// ---------------------------------------------------------------------------
__global__ __launch_bounds__(128)
void attn_fc(const float* __restrict__ H2, const float* __restrict__ aw_,
             const float* __restrict__ ab_, const float* __restrict__ fcw,
             const float* __restrict__ fcb, float* __restrict__ out)
{
    const int b = blockIdx.x;
    const int t = threadIdx.x;
    __shared__ float sc[SS];
    __shared__ float wgt[SS];
    __shared__ float ctx[128];
    __shared__ float mx_s, sum_s;

    const float* hb = H2 + (size_t)b * SS * 128;

    if (t < SS) {
        float s = ab_[0];
        for (int k = 0; k < 128; ++k) s = fmaf(hb[t * 128 + k], aw_[k], s);
        sc[t] = tanhf(s);
    }
    __syncthreads();
    if (t == 0) {
        float mx = sc[0];
        for (int i = 1; i < SS; ++i) mx = fmaxf(mx, sc[i]);
        mx_s = mx;
    }
    __syncthreads();
    if (t < SS) wgt[t] = __expf(sc[t] - mx_s);
    __syncthreads();
    if (t == 0) {
        float sm = 0.f;
        for (int i = 0; i < SS; ++i) sm += wgt[i];
        sum_s = sm;
    }
    __syncthreads();
    const float inv = 1.f / sum_s;
    float cj = 0.f;
    for (int tt = 0; tt < SS; ++tt) cj = fmaf(wgt[tt] * inv, hb[tt * 128 + t], cj);
    ctx[t] = cj;
    __syncthreads();
    if (t < NCLS) {
        float o = fcb[t];
        for (int k = 0; k < 128; ++k) o = fmaf(ctx[k], fcw[t * 128 + k], o);
        out[b * NCLS + t] = o;
    }
}

// ---------------------------------------------------------------------------
// Launch
// ---------------------------------------------------------------------------
extern "C" void kernel_launch(void* const* d_in, const int* in_sizes, int n_in,
                              void* d_out, int out_size, void* d_ws, size_t ws_size,
                              hipStream_t stream)
{
    (void)in_sizes; (void)n_in; (void)out_size; (void)ws_size;

    const float* x  = (const float*)d_in[0];
    const int*   ei = (const int*)d_in[1];
    const float *gw[4], *gas[4], *gad[4], *gb[4];
    for (int i = 0; i < 4; ++i) {
        gw[i]  = (const float*)d_in[2 + 4 * i];
        gas[i] = (const float*)d_in[3 + 4 * i];
        gad[i] = (const float*)d_in[4 + 4 * i];
        gb[i]  = (const float*)d_in[5 + 4 * i];
    }
    const float* wih0 = (const float*)d_in[18];
    const float* whh0 = (const float*)d_in[19];
    const float* bih0 = (const float*)d_in[20];
    const float* bhh0 = (const float*)d_in[21];
    const float* wih1 = (const float*)d_in[22];
    const float* whh1 = (const float*)d_in[23];
    const float* bih1 = (const float*)d_in[24];
    const float* bhh1 = (const float*)d_in[25];
    const float* attn_w = (const float*)d_in[26];
    const float* attn_b = (const float*)d_in[27];
    const float* fc_w   = (const float*)d_in[28];
    const float* fc_b   = (const float*)d_in[29];

    float* ws = (float*)d_ws;
    const size_t GNC = (size_t)GN * 128;  // 35,651,584 floats per GAT buffer
    float* bufT = ws;
    float* bufU = ws + GNC;
    // LSTM scratch overlaps bufT (free after the last GAT GEMM consumes it)
    float* X0 = bufT;                          // [16384, 512]
    float* H1 = bufT + (size_t)NG * 512;       // [16384, 128]
    float* X1 = H1   + (size_t)NG * 128;       // [16384, 512]
    float* H2 = X1   + (size_t)NG * 512;       // [16384, 128]

    const dim3 gblk(256);
    const dim3 ablk(128);
    const dim3 gat_grid(GN / 32, 1);           // 8704 blocks
    const dim3 lstm_grid(NG / 32, 4);          // 512 x 4 blocks

    // ---- GAT layer 0 (K=3), ReLU ----
    gemm_bf16_wmma<<<gat_grid, gblk, 0, stream>>>(x, CIN, gw[0], 128, 0,
        nullptr, nullptr, bufT, 128, GN, CIN, 128, 0);
    gat_aggregate<<<dim3(NG), ablk, 0, stream>>>(bufT, ei, gas[0], gad[0], gb[0], 1);
    // ---- GAT layer 1 ----
    gemm_bf16_wmma<<<gat_grid, gblk, 0, stream>>>(bufT, 128, gw[1], 128, 0,
        nullptr, nullptr, bufU, 128, GN, 128, 128, 0);
    gat_aggregate<<<dim3(NG), ablk, 0, stream>>>(bufU, ei, gas[1], gad[1], gb[1], 0);
    // ---- GAT layer 2, ReLU ----
    gemm_bf16_wmma<<<gat_grid, gblk, 0, stream>>>(bufU, 128, gw[2], 128, 0,
        nullptr, nullptr, bufT, 128, GN, 128, 128, 0);
    gat_aggregate<<<dim3(NG), ablk, 0, stream>>>(bufT, ei, gas[2], gad[2], gb[2], 1);
    // ---- GAT layer 3 ----
    gemm_bf16_wmma<<<gat_grid, gblk, 0, stream>>>(bufT, 128, gw[3], 128, 0,
        nullptr, nullptr, bufU, 128, GN, 128, 128, 0);
    gat_aggregate<<<dim3(NG), ablk, 0, stream>>>(bufU, ei, gas[3], gad[3], gb[3], 0);

    // ---- LSTM layer 0: hoisted input projection (big GEMM), then recurrence
    // bufU viewed as [16384, 2176] == lstm_in
    gemm_bf16_wmma<<<lstm_grid, gblk, 0, stream>>>(bufU, LSTM_IN, wih0, LSTM_IN, 1,
        bih0, bhh0, X0, 512, NG, LSTM_IN, 512, 0);
    lstm_recur<<<dim3(BB), ablk, 0, stream>>>(X0, whh0, H1, SS);

    // ---- LSTM layer 1 ----
    gemm_bf16_wmma<<<lstm_grid, gblk, 0, stream>>>(H1, 128, wih1, 128, 1,
        bih1, bhh1, X1, 512, NG, 128, 512, 0);
    lstm_recur<<<dim3(BB), ablk, 0, stream>>>(X1, whh1, H2, SS);

    // ---- attention pooling + FC ----
    attn_fc<<<dim3(BB), ablk, 0, stream>>>(H2, attn_w, attn_b, fc_w, fc_b,
                                           (float*)d_out);
}
